// CNNLSTMGNNModel_9285719294459
// MI455X (gfx1250) — matrix operand order repaired and checked by reading, not compiled, not run on hardware
//
#include <hip/hip_runtime.h>
#include <hip/hip_bf16.h>

typedef __attribute__((ext_vector_type(16))) _Float16 v16h;
typedef __attribute__((ext_vector_type(8)))  float    v8f;
typedef __attribute__((ext_vector_type(4)))  unsigned int u32x4;
typedef __attribute__((ext_vector_type(8)))  int      i32x8;
typedef __attribute__((ext_vector_type(4)))  int      i32x4;

#define NN     65536
#define HH     256
#define FEAT   64
#define FOURH  1024

// ---- workspace layout (bytes) ----
#define WS_WIH   0u                      // 1024*64  f16 = 131072
#define WS_WG    131072u                 // 256*256  f16 = 131072
#define WS_WFC1  262144u                 // 128*256  f16 = 65536
#define WS_BSUM  327680u                 // 1024     f32 = 4096
#define WS_HSA   524288u                 // N*256 f16 = 33554432
#define WS_HSB   (524288u + 33554432u)   // N*256 f16 = 33554432

union Frag16 { v16h v; _Float16 h[16]; uint4 q[2]; };
union H8     { uint4 u; _Float16 h[8]; };

__device__ __forceinline__ v8f wmma16(const Frag16& a, const Frag16& b, v8f c) {
    return __builtin_amdgcn_wmma_f32_16x16x32_f16(false, a.v, false, b.v,
                                                  (short)0, c, false, false);
}
__device__ __forceinline__ float sigf(float x)  { return 1.0f / (1.0f + __expf(-x)); }
__device__ __forceinline__ float tanhf_(float x){ return 1.0f - 2.0f / (__expf(2.0f * x) + 1.0f); }

// ---- Tensor Data Mover: 1-D bulk copy global -> LDS (data_size = 8B units) ----
// D# per CDNA5 ISA ch.8: group0 = {count, lds_addr, global_addr, type=2},
// group1 = {data_size=3 (8B), tensor_dim0 = tile_dim0 = nbytes/8, stride = dim0}.
// This toolchain exposes the 6-arg builtin: (g0, g1, g2, g3, g4, cpol).
__device__ __forceinline__ void tdm_load_to_lds(const void* lds_ptr,
                                                const void* global_ptr,
                                                unsigned nbytes) {
    const unsigned n8 = nbytes >> 3;                 // 8-byte elements (<= 65535)
    const unsigned lds_addr = (unsigned)(uintptr_t)lds_ptr;   // low 32b = LDS offset
    const unsigned long long ga = (unsigned long long)(uintptr_t)global_ptr;
    u32x4 g0;
    g0[0] = 1u;                                      // count=1, user descriptor
    g0[1] = lds_addr;                                // lds_addr [63:32]
    g0[2] = (unsigned)ga;                            // global_addr [95:64]
    g0[3] = (unsigned)((ga >> 32) & 0x1FFFFFFu) | (2u << 30);  // addr[56:32] | type=2
    i32x8 g1;
    g1[0] = 0x30000;                                 // data_size=3 (8B), no multicast
    g1[1] = (int)((n8 & 0xFFFFu) << 16);             // tensor_dim0[15:0]  -> bits 63:48
    g1[2] = (int)(((n8 >> 16) & 0xFFFFu) | (1u << 16)); // dim0[31:16], tensor_dim1=1
    g1[3] = (int)((n8 & 0xFFFFu) << 16);             // tile_dim0 -> bits 127:112
    g1[4] = 0;                                       // tile_dim1/2 unused
    g1[5] = (int)n8;                                 // tensor_dim0_stride[31:0]
    g1[6] = 0;
    g1[7] = 0;
    i32x4 z4 = {0, 0, 0, 0};
    i32x8 z8 = {0, 0, 0, 0, 0, 0, 0, 0};
    __builtin_amdgcn_tensor_load_to_lds(g0, g1, z4, z4, z8, 0);
}

// ---------------- prep: fp32 weights -> f16, fold LSTM biases ----------------
__global__ void k_prep(const float* __restrict__ Wih, const float* __restrict__ Wg,
                       const float* __restrict__ Wfc1, const float* __restrict__ bih,
                       const float* __restrict__ bhh,
                       _Float16* __restrict__ wih_h, _Float16* __restrict__ wg_h,
                       _Float16* __restrict__ wfc1_h, float* __restrict__ bsum) {
    int idx = blockIdx.x * 256 + threadIdx.x;       // 65536 threads
    if (idx < FOURH * FEAT) wih_h[idx]  = (_Float16)Wih[idx];
    if (idx < HH * HH)      wg_h[idx]   = (_Float16)Wg[idx];
    if (idx < 128 * HH)     wfc1_h[idx] = (_Float16)Wfc1[idx];
    if (idx < FOURH)        bsum[idx]   = bih[idx] + bhh[idx];
}

// ---------------- fused CNN + single-step LSTM (WMMA gates GEMM) ----------------
__global__ __launch_bounds__(256) void k_cnn_lstm(
        const float* __restrict__ x, const float* __restrict__ W1,
        const float* __restrict__ b1, const float* __restrict__ W2,
        const float* __restrict__ b2, const _Float16* __restrict__ wih_h,
        const float* __restrict__ bsum, _Float16* __restrict__ hs) {
    __shared__ float sx[128 * 10];
    __shared__ float sh1[128 * 80];
    __shared__ __align__(16) _Float16 sfeat[128 * 64];
    __shared__ __align__(16) _Float16 sWih[FOURH * FEAT];   // 128 KB weight stage

    const int tid = threadIdx.x;
    const int node_base = blockIdx.x * 128;

    // kick off weight DMA (wave 0) -- overlaps with the CNN stages below
    if (tid < 32) tdm_load_to_lds(sWih, wih_h, FOURH * FEAT * 2);

    // stage 0: x tile -> LDS
    for (int i = tid; i < 128 * 10; i += 256) sx[i] = x[node_base * 10 + i];
    __syncthreads();

    // stage 1: conv1(16,k3,pad1) -> relu -> maxpool2 : h1[128][16][5]
    for (int i = tid; i < 128 * 80; i += 256) {
        int node = i / 80, rem = i - node * 80;
        int ch = rem / 5, q = rem - ch * 5;
        float w0 = W1[ch * 3], w1 = W1[ch * 3 + 1], w2 = W1[ch * 3 + 2], bb = b1[ch];
        const float* xr = &sx[node * 10];
        int p0 = 2 * q, p1 = 2 * q + 1;
        float o0 = bb + (p0 > 0 ? w0 * xr[p0 - 1] : 0.0f) + w1 * xr[p0] + w2 * xr[p0 + 1];
        float o1 = bb + w0 * xr[p0] + w1 * xr[p1] + (p1 < 9 ? w2 * xr[p1 + 1] : 0.0f);
        sh1[node * 80 + ch * 5 + q] = fmaxf(fmaxf(o0, o1), 0.0f);
    }
    __syncthreads();

    // stage 2: conv2(32,16,k3,pad1) -> relu -> pool(first 4) : feat[128][64] f16
    for (int i = tid; i < 128 * 64; i += 256) {
        int node = i >> 6, f = i & 63;
        int oc = f >> 1, q = f & 1;
        int p0 = 2 * q;
        float a0 = b2[oc], a1 = b2[oc];
        const float* hr = &sh1[node * 80];
        #pragma unroll
        for (int ic = 0; ic < 16; ++ic) {
            const float* wr = &W2[(oc * 16 + ic) * 3];
            const float* hc = &hr[ic * 5];
            #pragma unroll
            for (int k = 0; k < 3; ++k) {
                int t0 = p0 - 1 + k;
                float w = wr[k];
                a0 += w * (t0 >= 0 ? hc[t0] : 0.0f);
                a1 += w * hc[t0 + 1];
            }
        }
        sfeat[node * 64 + f] = (_Float16)fmaxf(fmaxf(a0, a1), 0.0f);
    }
    if (tid < 32) __builtin_amdgcn_s_wait_tensorcnt(0);  // weights landed
    __syncthreads();

    // stage 3: gates = feat @ W_ih^T (i,g,o only), fused LSTM activations
    const int wv = tid >> 5, lane = tid & 31;
    const int ln = lane & 15, hlf = lane >> 4;
    const int lr0 = wv * 16, row0 = node_base + lr0;

    Frag16 a0, a1;   // K tiles 0..31, 32..63
    a0.q[0] = *(const uint4*)&sfeat[(lr0 + ln) * 64 + 0  + hlf * 8];
    a0.q[1] = *(const uint4*)&sfeat[(lr0 + ln) * 64 + 16 + hlf * 8];
    a1.q[0] = *(const uint4*)&sfeat[(lr0 + ln) * 64 + 32 + hlf * 8];
    a1.q[1] = *(const uint4*)&sfeat[(lr0 + ln) * 64 + 48 + hlf * 8];

    #pragma unroll 4
    for (int ct = 0; ct < 16; ++ct) {
        const int c0 = ct * 16;
        v8f ci = {}, cg = {}, co = {};
        const int gbase[3] = {0, 512, 768};
        v8f* accs[3] = {&ci, &cg, &co};
        #pragma unroll
        for (int g = 0; g < 3; ++g) {
            const _Float16* wr = sWih + (gbase[g] + c0 + ln) * 64;
            Frag16 b;
            b.q[0] = *(const uint4*)(wr + hlf * 8);
            b.q[1] = *(const uint4*)(wr + 16 + hlf * 8);
            *accs[g] = wmma16(a0, b, *accs[g]);
            b.q[0] = *(const uint4*)(wr + 32 + hlf * 8);
            b.q[1] = *(const uint4*)(wr + 32 + 16 + hlf * 8);
            *accs[g] = wmma16(a1, b, *accs[g]);
        }
        const float bi = bsum[c0 + ln], bg = bsum[512 + c0 + ln], bo = bsum[768 + c0 + ln];
        #pragma unroll
        for (int r = 0; r < 8; ++r) {
            float cc = sigf(ci[r] + bi) * tanhf_(cg[r] + bg);
            float hv = sigf(co[r] + bo) * tanhf_(cc);
            hs[(size_t)(row0 + r + hlf * 8) * HH + c0 + ln] = (_Float16)hv;
        }
    }
}

// ---------------- GCN GEMM: m = hs @ Wg^T (16x256 per wave, K=256) ----------------
__global__ __launch_bounds__(256) void k_gcn_gemm(const _Float16* __restrict__ hs,
                                                  const _Float16* __restrict__ wg_h,
                                                  _Float16* __restrict__ m) {
    __shared__ __align__(16) _Float16 sW[HH * HH];   // 128 KB weight stage
    const int tid = threadIdx.x, wv = tid >> 5, lane = tid & 31;
    const int ln = lane & 15, hlf = lane >> 4;
    const int row0 = blockIdx.x * 128 + wv * 16;

    if (tid < 32) {
        tdm_load_to_lds(sW, wg_h, HH * HH * 2);
        __builtin_amdgcn_s_wait_tensorcnt(0);
    }

    Frag16 A[8];
    #pragma unroll
    for (int kt = 0; kt < 8; ++kt) {
        const _Float16* ar = hs + (size_t)(row0 + ln) * HH + kt * 32;
        A[kt].q[0] = *(const uint4*)(ar + hlf * 8);
        A[kt].q[1] = *(const uint4*)(ar + 16 + hlf * 8);
    }
    __syncthreads();

    #pragma unroll 2
    for (int ct = 0; ct < 16; ++ct) {
        const int c0 = ct * 16;
        v8f acc = {};
        const _Float16* wr = sW + (c0 + ln) * HH;
        #pragma unroll
        for (int kt = 0; kt < 8; ++kt) {
            Frag16 b;
            b.q[0] = *(const uint4*)(wr + kt * 32 + hlf * 8);
            b.q[1] = *(const uint4*)(wr + kt * 32 + 16 + hlf * 8);
            acc = wmma16(A[kt], b, acc);
        }
        #pragma unroll
        for (int r = 0; r < 8; ++r)
            m[(size_t)(row0 + r + hlf * 8) * HH + c0 + ln] = (_Float16)acc[r];
    }
}

// ---------------- path-graph GCN stencil: out = relu(sum norm*m + bg) ----------------
__global__ __launch_bounds__(256) void k_stencil(const _Float16* __restrict__ m,
                                                 const float* __restrict__ bg,
                                                 _Float16* __restrict__ out) {
    const int gid = blockIdx.x * 256 + threadIdx.x;     // N*32 threads
    const int row = gid >> 5;
    const int c0 = (gid & 31) * 8;
    const float DI = 0.57735026919f;     // 1/sqrt(3) interior
    const float DE = 0.70710678119f;     // 1/sqrt(2) endpoints
    const float di = (row == 0 || row == NN - 1) ? DE : DI;

    float acc[8];
    H8 v;
    v.u = *(const uint4*)(m + (size_t)row * HH + c0);
    const float cs = di * di;
    #pragma unroll
    for (int k = 0; k < 8; ++k) acc[k] = cs * (float)v.h[k];
    if (row > 0) {
        const float dj = (row - 1 == 0) ? DE : DI;
        const float cf = di * dj;
        v.u = *(const uint4*)(m + (size_t)(row - 1) * HH + c0);
        #pragma unroll
        for (int k = 0; k < 8; ++k) acc[k] += cf * (float)v.h[k];
    }
    if (row < NN - 1) {
        const float dj = (row + 1 == NN - 1) ? DE : DI;
        const float cf = di * dj;
        v.u = *(const uint4*)(m + (size_t)(row + 1) * HH + c0);
        #pragma unroll
        for (int k = 0; k < 8; ++k) acc[k] += cf * (float)v.h[k];
    }
    H8 o;
    #pragma unroll
    for (int k = 0; k < 8; ++k) o.h[k] = (_Float16)fmaxf(acc[k] + bg[c0 + k], 0.0f);
    *(uint4*)(out + (size_t)row * HH + c0) = o.u;
}

// ---------------- head: relu(hs@Wfc1^T+b) -> @Wfc2^T+b -> log_softmax ----------------
__global__ __launch_bounds__(256) void k_head(const _Float16* __restrict__ hs,
                                              const _Float16* __restrict__ wfc1_h,
                                              const float* __restrict__ bfc1,
                                              const float* __restrict__ Wfc2,
                                              const float* __restrict__ bfc2,
                                              float* __restrict__ out) {
    __shared__ float z1[8][16][132];                  // padded: conflict-free tail reads
    __shared__ __align__(16) _Float16 sW[128 * HH];   // 64 KB weight stage
    const int tid = threadIdx.x, wv = tid >> 5, lane = tid & 31;
    const int ln = lane & 15, hlf = lane >> 4;
    const int row0 = blockIdx.x * 128 + wv * 16;

    if (tid < 32) {
        tdm_load_to_lds(sW, wfc1_h, 128 * HH * 2);
        __builtin_amdgcn_s_wait_tensorcnt(0);
    }

    Frag16 A[8];
    #pragma unroll
    for (int kt = 0; kt < 8; ++kt) {
        const _Float16* ar = hs + (size_t)(row0 + ln) * HH + kt * 32;
        A[kt].q[0] = *(const uint4*)(ar + hlf * 8);
        A[kt].q[1] = *(const uint4*)(ar + 16 + hlf * 8);
    }
    __syncthreads();

    #pragma unroll 2
    for (int ct = 0; ct < 8; ++ct) {
        const int c0 = ct * 16;
        v8f acc = {};
        const _Float16* wr = sW + (c0 + ln) * HH;
        #pragma unroll
        for (int kt = 0; kt < 8; ++kt) {
            Frag16 b;
            b.q[0] = *(const uint4*)(wr + kt * 32 + hlf * 8);
            b.q[1] = *(const uint4*)(wr + kt * 32 + 16 + hlf * 8);
            acc = wmma16(A[kt], b, acc);
        }
        const float bb = bfc1[c0 + ln];
        #pragma unroll
        for (int r = 0; r < 8; ++r)
            z1[wv][r + hlf * 8][c0 + ln] = fmaxf(acc[r] + bb, 0.0f);
    }
    // per-row 128 -> 3 dot + log_softmax (same-wave DS ops are in-order)
    if (lane < 16) {
        const float* zr = z1[wv][lane];
        float z[3];
        #pragma unroll
        for (int t = 0; t < 3; ++t) {
            float s = bfc2[t];
            const float* w2 = Wfc2 + t * 128;
            for (int c = 0; c < 128; ++c) s += zr[c] * w2[c];
            z[t] = s;
        }
        float mx = fmaxf(fmaxf(z[0], z[1]), z[2]);
        float lse = mx + __logf(__expf(z[0] - mx) + __expf(z[1] - mx) + __expf(z[2] - mx));
        const size_t ob = (size_t)(row0 + lane) * 3;
        out[ob + 0] = z[0] - lse;
        out[ob + 1] = z[1] - lse;
        out[ob + 2] = z[2] - lse;
    }
}

extern "C" void kernel_launch(void* const* d_in, const int* in_sizes, int n_in,
                              void* d_out, int out_size, void* d_ws, size_t ws_size,
                              hipStream_t stream) {
    (void)in_sizes; (void)n_in; (void)out_size; (void)ws_size;
    const float* x    = (const float*)d_in[0];
    const float* W1   = (const float*)d_in[1];
    const float* b1   = (const float*)d_in[2];
    const float* W2   = (const float*)d_in[3];
    const float* b2   = (const float*)d_in[4];
    const float* Wih  = (const float*)d_in[5];
    const float* bih  = (const float*)d_in[7];
    const float* bhh  = (const float*)d_in[8];
    const float* Wg   = (const float*)d_in[9];
    const float* bg   = (const float*)d_in[10];
    const float* Wfc1 = (const float*)d_in[12];
    const float* bfc1 = (const float*)d_in[13];
    const float* Wfc2 = (const float*)d_in[14];
    const float* bfc2 = (const float*)d_in[15];
    float* out = (float*)d_out;

    char* ws = (char*)d_ws;
    _Float16* wih_h  = (_Float16*)(ws + WS_WIH);
    _Float16* wg_h   = (_Float16*)(ws + WS_WG);
    _Float16* wfc1_h = (_Float16*)(ws + WS_WFC1);
    float*    bsum   = (float*)   (ws + WS_BSUM);
    _Float16* hsA    = (_Float16*)(ws + WS_HSA);
    _Float16* hsB    = (_Float16*)(ws + WS_HSB);

    k_prep<<<256, 256, 0, stream>>>(Wih, Wg, Wfc1, bih, bhh, wih_h, wg_h, wfc1_h, bsum);
    k_cnn_lstm<<<NN / 128, 256, 0, stream>>>(x, W1, b1, W2, b2, wih_h, bsum, hsA);
    for (int layer = 0; layer < 2; ++layer) {
        k_gcn_gemm<<<NN / 128, 256, 0, stream>>>(hsA, wg_h, hsB);
        k_stencil<<<NN * 32 / 256, 256, 0, stream>>>(hsB, bg, hsA);
    }
    k_head<<<NN / 128, 256, 0, stream>>>(hsA, wfc1_h, bfc1, Wfc2, bfc2, out);
}
